// TemporalMambaBlock_46677704573691
// MI455X (gfx1250) — matrix operand order repaired and checked by reading, not compiled, not run on hardware
//
#include <hip/hip_runtime.h>
#include <hip/hip_bf16.h>
#include <math.h>

// ---------------------------------------------------------------------------
// Problem constants (match reference)
// ---------------------------------------------------------------------------
#define BB      2
#define TT      16
#define CC      128
#define HH      32
#define WW      32
#define DSTATE  16
#define DCONV   4
#define DINNER  256
#define DTRANK  8
#define HIDDEN  512
#define NSEQ    (BB * HH * WW)      // 2048
#define MTOK    (NSEQ * TT)         // 32768
#define EPS     1e-6f

// strides of x [B,T,C,H,W]
#define XSB  (TT * CC * HH * WW)    // 2097152
#define XST  (CC * HH * WW)         // 131072
#define XSC  (HH * WW)              // 1024

typedef __attribute__((ext_vector_type(16))) _Float16 v16h;
typedef __attribute__((ext_vector_type(8)))  float    v8f;
typedef __attribute__((ext_vector_type(8)))  unsigned  v8u;

// ---------------------------------------------------------------------------
// WMMA fragment helpers (16x16x32 f16 -> f32), wave32 layouts per CDNA5 ISA.
// A 16x32 (MxK):  lanes 0-15: M=lane, VGPR j -> K=(j<4?0:16)+(j&3)*2 (+8 for
// lanes 16-31).  B 32x16 (KxN): N=lane&15, same K mapping as A (VGPR j ->
// K=(j<4?0:16)+(j&3)*2, +8 for lanes 16-31).  C/D f32: VGPR r -> M=r+8*(lane>>4), N=lane&15.
// All K offsets are even so each fragment element is one b32 load (the
// compiler further coalesces these into b128 clauses).
// ---------------------------------------------------------------------------
__device__ __forceinline__ v16h load_frag_a(const _Float16* base, int ld) {
  int lane = threadIdx.x & 31;
  int half = lane >> 4, row = lane & 15;
  const _Float16* p = base + row * ld + (half << 3);
  v8u u;
#pragma unroll
  for (int j = 0; j < 8; ++j) {
    int k = ((j & 4) << 2) + ((j & 3) << 1);   // {0,2,4,6,16,18,20,22}
    u[j] = *(const unsigned*)(p + k);
  }
  return __builtin_bit_cast(v16h, u);
}

// wt points at W^T stored row-major [N][ldk] (i.e. W[n][k], k contiguous)
__device__ __forceinline__ v16h load_frag_b(const _Float16* wt, int ldk) {
  int lane = threadIdx.x & 31;
  int n = lane & 15, kg = (lane >> 4) << 3;
  const _Float16* p = wt + n * ldk + kg;
  v8u u;
#pragma unroll
  for (int j = 0; j < 8; ++j) u[j] = *(const unsigned*)(p + ((j & 4) << 2) + ((j & 3) << 1));
  return __builtin_bit_cast(v16h, u);
}

__device__ __forceinline__ v8f wmma16(v16h a, v16h b, v8f c) {
  return __builtin_amdgcn_wmma_f32_16x16x32_f16(
      false, a, false, b, (short)0, c, false, false);
}

__device__ __forceinline__ float siluf(float x) {
  return x / (1.0f + expf(-x));
}

// ---------------------------------------------------------------------------
// Kernel 0: weight conversion fp32 -> f16 (with zero padding) + A = -exp(Alog)
// ---------------------------------------------------------------------------
#define N_IPW  (2 * DINNER * CC)          // 65536
#define N_XPW  (48 * DINNER)              // 12288 (rows 40..47 zero)
#define N_DTW  (DINNER * 32)              // 8192  (cols 8..31 zero)
#define N_OPW  (CC * DINNER)              // 32768
#define N_F1W  (HIDDEN * CC)              // 65536
#define N_F2W  (CC * HIDDEN)              // 65536
#define N_ANEG (DINNER * DSTATE)          // 4096
#define PREP_TOTAL (N_IPW+N_XPW+N_DTW+N_OPW+N_F1W+N_F2W+N_ANEG)

__global__ void k_prep(const float* ipw32, const float* xpw32,
                       const float* dtw32, const float* opw32,
                       const float* f1w32, const float* f2w32,
                       const float* A_log,
                       _Float16* ipw, _Float16* xpw, _Float16* dtw,
                       _Float16* opw, _Float16* f1w, _Float16* f2w,
                       float* Aneg) {
  for (int i = blockIdx.x * blockDim.x + threadIdx.x; i < PREP_TOTAL;
       i += gridDim.x * blockDim.x) {
    int j = i;
    if (j < N_IPW) { ipw[j] = (_Float16)ipw32[j]; continue; }
    j -= N_IPW;
    if (j < N_XPW) {
      int r = j / DINNER, c = j % DINNER;
      xpw[j] = (r < DTRANK + 2 * DSTATE) ? (_Float16)xpw32[r * DINNER + c]
                                         : (_Float16)0.0f;
      continue;
    }
    j -= N_XPW;
    if (j < N_DTW) {
      int r = j >> 5, c = j & 31;
      dtw[j] = (c < DTRANK) ? (_Float16)dtw32[r * DTRANK + c] : (_Float16)0.0f;
      continue;
    }
    j -= N_DTW;
    if (j < N_OPW) { opw[j] = (_Float16)opw32[j]; continue; }
    j -= N_OPW;
    if (j < N_F1W) { f1w[j] = (_Float16)f1w32[j]; continue; }
    j -= N_F1W;
    if (j < N_F2W) { f2w[j] = (_Float16)f2w32[j]; continue; }
    j -= N_F2W;
    Aneg[j] = -expf(A_log[j]);
  }
}

// ---------------------------------------------------------------------------
// Kernel 1: gather [B,T,C,H,W] -> [M,128], RMSNorm -> xn (f16) + residual xt
// 256 threads / block, 16 threads per token (8 channels each), blockIdx = seq
// ---------------------------------------------------------------------------
__global__ void k_gather_norm(const float* __restrict__ x,
                              const float* __restrict__ norm1_w,
                              float* __restrict__ xt,
                              _Float16* __restrict__ xn) {
  int seq = blockIdx.x;
  int tid = threadIdx.x;
  int t = tid >> 4, i = tid & 15;
  int b = seq >> 10, hw = seq & 1023;
  const float* src = x + (size_t)b * XSB + (size_t)t * XST + hw;
  size_t m = (size_t)seq * TT + t;
  float v[8];
  float ss = 0.0f;
#pragma unroll
  for (int j = 0; j < 8; ++j) {
    int c = i * 8 + j;
    v[j] = src[(size_t)c * XSC];
    ss += v[j] * v[j];
  }
#pragma unroll
  for (int off = 8; off; off >>= 1) ss += __shfl_xor(ss, off, 16);
  float scale = rsqrtf(ss * (1.0f / CC) + EPS);
#pragma unroll
  for (int j = 0; j < 8; ++j) {
    int c = i * 8 + j;
    xt[m * CC + c] = v[j];
    xn[m * CC + c] = (_Float16)(v[j] * scale * norm1_w[c]);
  }
}

// ---------------------------------------------------------------------------
// Kernel 2: in_proj GEMM  xn[M,128] x W^T[128,512] -> xi[M,256], z[M,256]
// 8 waves/block, one 16x16 output tile per wave, K loop of 4 (K=128).
// The xi/z split is decided by n_tile (wave-uniform) so the store epilogue
// has no per-lane divergence.
// ---------------------------------------------------------------------------
__global__ void k_in_proj(const _Float16* __restrict__ xn,
                          const _Float16* __restrict__ ipw,
                          _Float16* __restrict__ xi,
                          _Float16* __restrict__ z) {
  int wave = (blockIdx.x * blockDim.x + threadIdx.x) >> 5;
  int m_tile = wave >> 5;           // 2048 m-tiles
  int n_tile = wave & 31;           // 32 n-tiles
  int lane = threadIdx.x & 31;
  const _Float16* abase = xn + (size_t)m_tile * 16 * CC;
  const _Float16* bbase = ipw + (size_t)n_tile * 16 * CC;
  v8f acc = {};
#pragma unroll
  for (int kk = 0; kk < 4; ++kk) {
    v16h a = load_frag_a(abase + kk * 32, CC);
    v16h b = load_frag_b(bbase + kk * 32, CC);
    acc = wmma16(a, b, acc);
  }
  int col = lane & 15, rb = (lane >> 4) << 3;
  // wave-uniform destination select: n_tile<16 -> xi, else -> z
  _Float16* dst = (n_tile < 16) ? xi : z;
  int e = ((n_tile & 15) << 4) + col;
  dst += (size_t)(m_tile * 16 + rb) * DINNER + e;
#pragma unroll
  for (int r = 0; r < 8; ++r)
    dst[(size_t)r * DINNER] = (_Float16)acc[r];
}

// ---------------------------------------------------------------------------
// Kernel 3: per-sequence fused mamba (fwd + bwd dirs in one workgroup)
// conv+silu -> x_proj (WMMA) -> dt_proj (WMMA, K padded) -> softplus ->
// selective scan (4 waves x 64 channels, 2 ch/lane, 16 states in regs) ->
// silu(z) gate -> out_proj (WMMA) accumulated onto residual in LDS.
// 128 threads (4 waves), blockIdx = seq.
// ---------------------------------------------------------------------------
#define LDXC 264   // padded leading dim (even) for 16x256 f16 LDS tiles

__global__ void k_mamba(const _Float16* __restrict__ xi_g,
                        const _Float16* __restrict__ z_g,
                        const float* __restrict__ conv_w,
                        const float* __restrict__ conv_b,
                        const _Float16* __restrict__ xpw,
                        const _Float16* __restrict__ dtw,
                        const float* __restrict__ dt_proj_b,
                        const float* __restrict__ Aneg,
                        const float* __restrict__ D_param,
                        const _Float16* __restrict__ opw,
                        const float* __restrict__ xt,
                        float* __restrict__ xmid) {
  __shared__ _Float16 sXI[TT * DINNER];
  __shared__ _Float16 sXC[TT * LDXC];
  __shared__ float    sDBL[TT * 48];
  __shared__ _Float16 sDT[TT * 32];
  __shared__ float    sDTV[TT * DINNER];
  __shared__ _Float16 sY[TT * LDXC];
  __shared__ float    sOUT[TT * CC];

  int seq = blockIdx.x;
  int tid = threadIdx.x;            // 128
  int wave = tid >> 5, lane = tid & 31;
  size_t base256 = (size_t)seq * TT * DINNER;
  size_t base128 = (size_t)seq * TT * CC;

  // stage xi tile + residual into LDS (dword copies)
  for (int o = tid; o < TT * DINNER / 2; o += 128)
    ((unsigned*)sXI)[o] = ((const unsigned*)(xi_g + base256))[o];
  for (int o = tid; o < TT * CC; o += 128) sOUT[o] = xt[base128 + o];

  // per-lane scan constants: lane owns channels d0, d0+1
  int d0 = wave * 64 + lane * 2;
  float An[2][DSTATE], Dp[2];
#pragma unroll
  for (int dd = 0; dd < 2; ++dd) {
    Dp[dd] = D_param[d0 + dd];
#pragma unroll
    for (int s = 0; s < DSTATE; ++s) An[dd][s] = Aneg[(d0 + dd) * DSTATE + s];
  }
  __syncthreads();

  for (int dir = 0; dir < 2; ++dir) {
    // ---- causal depthwise conv (on time-flipped stream for dir=1) + SiLU
    for (int o = tid; o < TT * DINNER; o += 128) {
      int t = o >> 8, d = o & 255;
      float acc = conv_b[d];
#pragma unroll
      for (int k = 0; k < DCONV; ++k) {
        int ts = t - (DCONV - 1) + k;
        if (ts >= 0) {
          int tr = dir ? (TT - 1 - ts) : ts;
          acc += (float)sXI[tr * DINNER + d] * conv_w[d * DCONV + k];
        }
      }
      sXC[t * LDXC + d] = (_Float16)siluf(acc);
    }
    __syncthreads();

    // ---- x_proj: [16x256] x [256x48pad] -> dbl[16x40]; waves 0..2
    if (wave < 3) {
      v8f acc = {};
#pragma unroll
      for (int kk = 0; kk < 8; ++kk) {
        v16h a = load_frag_a(sXC + kk * 32, LDXC);
        v16h b = load_frag_b(xpw + wave * 16 * DINNER + kk * 32, DINNER);
        acc = wmma16(a, b, acc);
      }
      int col = lane & 15, rb = (lane >> 4) << 3;
      int e = wave * 16 + col;
      if (e < DTRANK + 2 * DSTATE)
#pragma unroll
        for (int r = 0; r < 8; ++r) sDBL[(rb + r) * 48 + e] = acc[r];
    }
    __syncthreads();

    // ---- build K-padded dt A-matrix [16x32]
    for (int o = tid; o < TT * 32; o += 128) {
      int t = o >> 5, e = o & 31;
      sDT[o] = (e < DTRANK) ? (_Float16)sDBL[t * 48 + e] : (_Float16)0.0f;
    }
    __syncthreads();

    // ---- dt_proj: [16x32pad] x [32pad x 256] + bias -> softplus -> sDTV
    {
      v16h a = load_frag_a(sDT, 32);
      for (int nt = wave; nt < 16; nt += 4) {
        v16h b = load_frag_b(dtw + nt * 16 * 32, 32);
        v8f acc = {};
        acc = wmma16(a, b, acc);
        int col = lane & 15, rb = (lane >> 4) << 3;
        int d = nt * 16 + col;
#pragma unroll
        for (int r = 0; r < 8; ++r) {
          float v = acc[r] + dt_proj_b[d];
          sDTV[(rb + r) * DINNER + d] = (v > 20.0f) ? v : log1pf(expf(v));
        }
      }
    }
    __syncthreads();

    // ---- selective scan; write gated y at ORIGINAL time row for out_proj
    {
      float h[2][DSTATE];
#pragma unroll
      for (int dd = 0; dd < 2; ++dd)
#pragma unroll
        for (int s = 0; s < DSTATE; ++s) h[dd][s] = 0.0f;

      for (int t = 0; t < TT; ++t) {
        int t_orig = dir ? (TT - 1 - t) : t;
        float Bv[DSTATE], Cv[DSTATE];
#pragma unroll
        for (int s = 0; s < DSTATE; ++s) {
          Bv[s] = sDBL[t * 48 + DTRANK + s];
          Cv[s] = sDBL[t * 48 + DTRANK + DSTATE + s];
        }
#pragma unroll
        for (int dd = 0; dd < 2; ++dd) {
          int d = d0 + dd;
          float dtv = sDTV[t * DINNER + d];
          float u = (float)sXC[t * LDXC + d];
          float y = 0.0f;
#pragma unroll
          for (int s = 0; s < DSTATE; ++s) {
            float hh = h[dd][s] * __expf(dtv * An[dd][s]) + dtv * Bv[s] * u;
            h[dd][s] = hh;
            y += hh * Cv[s];
          }
          y += Dp[dd] * u;
          float zv = (float)z_g[base256 + (size_t)t_orig * DINNER + d];
          sY[t_orig * LDXC + d] = (_Float16)(y * siluf(zv));
        }
      }
    }
    __syncthreads();

    // ---- out_proj: [16x256] x [256x128] accumulated into residual
    for (int nt = wave; nt < 8; nt += 4) {
      v8f acc = {};
#pragma unroll
      for (int kk = 0; kk < 8; ++kk) {
        v16h a = load_frag_a(sY + kk * 32, LDXC);
        v16h b = load_frag_b(opw + nt * 16 * DINNER + kk * 32, DINNER);
        acc = wmma16(a, b, acc);
      }
      int col = lane & 15, rb = (lane >> 4) << 3;
      int c = nt * 16 + col;
#pragma unroll
      for (int r = 0; r < 8; ++r) sOUT[(rb + r) * CC + c] += acc[r];
    }
    __syncthreads();
  }

  for (int o = tid; o < TT * CC; o += 128) xmid[base128 + o] = sOUT[o];
}

// ---------------------------------------------------------------------------
// Kernel 4: RMSNorm -> fc1 (WMMA) + GELU -> fc2 (WMMA) + bias + residual,
// scatter back to [B,T,C,H,W].  128 threads (4 waves), blockIdx = seq.
// ---------------------------------------------------------------------------
#define LDH 520

__global__ void k_mlp(const float* __restrict__ xmid,
                      const float* __restrict__ norm2_w,
                      const _Float16* __restrict__ f1w,
                      const float* __restrict__ fc1_b,
                      const _Float16* __restrict__ f2w,
                      const float* __restrict__ fc2_b,
                      float* __restrict__ out) {
  __shared__ _Float16 sXN[TT * CC];
  __shared__ _Float16 sH[TT * LDH];
  __shared__ float    sXM[TT * CC];

  int seq = blockIdx.x, tid = threadIdx.x;
  int wave = tid >> 5, lane = tid & 31;
  size_t base128 = (size_t)seq * TT * CC;

  // load + RMSNorm: 8 threads per row, 16 channels each
  {
    int t = tid >> 3, i = tid & 7;
    float v[16];
    float ss = 0.0f;
#pragma unroll
    for (int j = 0; j < 16; ++j) {
      v[j] = xmid[base128 + t * CC + i * 16 + j];
      ss += v[j] * v[j];
    }
#pragma unroll
    for (int off = 4; off; off >>= 1) ss += __shfl_xor(ss, off, 8);
    float scale = rsqrtf(ss * (1.0f / CC) + EPS);
#pragma unroll
    for (int j = 0; j < 16; ++j) {
      int c = i * 16 + j;
      sXM[t * CC + c] = v[j];
      sXN[t * CC + c] = (_Float16)(v[j] * scale * norm2_w[c]);
    }
  }
  __syncthreads();

  // fc1 + exact GELU: 32 n-tiles, 8 per wave
  for (int nt = wave; nt < 32; nt += 4) {
    v8f acc = {};
#pragma unroll
    for (int kk = 0; kk < 4; ++kk) {
      v16h a = load_frag_a(sXN + kk * 32, CC);
      v16h b = load_frag_b(f1w + nt * 16 * CC + kk * 32, CC);
      acc = wmma16(a, b, acc);
    }
    int col = lane & 15, rb = (lane >> 4) << 3;
    int hc = nt * 16 + col;
#pragma unroll
    for (int r = 0; r < 8; ++r) {
      float v = acc[r] + fc1_b[hc];
      float g = 0.5f * v * (1.0f + erff(v * 0.70710678118f));
      sH[(rb + r) * LDH + hc] = (_Float16)g;
    }
  }
  __syncthreads();

  // fc2 + bias + residual, scatter to output layout
  int b = seq >> 10, hw = seq & 1023;
  for (int nt = wave; nt < 8; nt += 4) {
    v8f acc = {};
#pragma unroll
    for (int kk = 0; kk < 16; ++kk) {
      v16h a = load_frag_a(sH + kk * 32, LDH);
      v16h bf = load_frag_b(f2w + nt * 16 * HIDDEN + kk * 32, HIDDEN);
      acc = wmma16(a, bf, acc);
    }
    int col = lane & 15, rb = (lane >> 4) << 3;
    int c = nt * 16 + col;
#pragma unroll
    for (int r = 0; r < 8; ++r) {
      int t = rb + r;
      float v = acc[r] + fc2_b[c] + sXM[t * CC + c];
      out[(size_t)b * XSB + (size_t)t * XST + (size_t)c * XSC + hw] = v;
    }
  }
}

// ---------------------------------------------------------------------------
// Host side
// ---------------------------------------------------------------------------
extern "C" void kernel_launch(void* const* d_in, const int* in_sizes, int n_in,
                              void* d_out, int out_size, void* d_ws, size_t ws_size,
                              hipStream_t stream) {
  const float* x         = (const float*)d_in[0];
  const float* norm1_w   = (const float*)d_in[1];
  const float* in_proj_w = (const float*)d_in[2];
  const float* conv_w    = (const float*)d_in[3];
  const float* conv_b    = (const float*)d_in[4];
  const float* x_proj_w  = (const float*)d_in[5];
  const float* dt_proj_w = (const float*)d_in[6];
  const float* dt_proj_b = (const float*)d_in[7];
  const float* A_log     = (const float*)d_in[8];
  const float* D_param   = (const float*)d_in[9];
  const float* out_proj_w= (const float*)d_in[10];
  const float* norm2_w   = (const float*)d_in[11];
  const float* fc1_w     = (const float*)d_in[12];
  const float* fc1_b     = (const float*)d_in[13];
  const float* fc2_w     = (const float*)d_in[14];
  const float* fc2_b     = (const float*)d_in[15];
  float* out = (float*)d_out;

  char* ws = (char*)d_ws;
  size_t off = 0;
  _Float16* ipw  = (_Float16*)(ws + off); off += (size_t)N_IPW * 2;   // 131072
  _Float16* xpw  = (_Float16*)(ws + off); off += (size_t)N_XPW * 2;   //  24576
  _Float16* dtw  = (_Float16*)(ws + off); off += (size_t)N_DTW * 2;   //  16384
  _Float16* opw  = (_Float16*)(ws + off); off += (size_t)N_OPW * 2;   //  65536
  _Float16* f1w  = (_Float16*)(ws + off); off += (size_t)N_F1W * 2;   // 131072
  _Float16* f2w  = (_Float16*)(ws + off); off += (size_t)N_F2W * 2;   // 131072
  float*    Aneg = (float*)   (ws + off); off += (size_t)N_ANEG * 4;  //  16384
  float*    xt   = (float*)   (ws + off); off += (size_t)MTOK * CC * 4;
  _Float16* xn   = (_Float16*)(ws + off); off += (size_t)MTOK * CC * 2;
  _Float16* xi   = (_Float16*)(ws + off); off += (size_t)MTOK * DINNER * 2;
  _Float16* z    = (_Float16*)(ws + off); off += (size_t)MTOK * DINNER * 2;
  float*    xmid = (float*)   (ws + off); off += (size_t)MTOK * CC * 4;

  k_prep<<<128, 256, 0, stream>>>(in_proj_w, x_proj_w, dt_proj_w, out_proj_w,
                                  fc1_w, fc2_w, A_log,
                                  ipw, xpw, dtw, opw, f1w, f2w, Aneg);
  k_gather_norm<<<NSEQ, 256, 0, stream>>>(x, norm1_w, xt, xn);
  k_in_proj<<<(MTOK / 16) * 32 / 8, 256, 0, stream>>>(xn, ipw, xi, z);
  k_mamba<<<NSEQ, 128, 0, stream>>>(xi, z, conv_w, conv_b, xpw, dtw,
                                    dt_proj_b, Aneg, D_param, opw, xt, xmid);
  k_mlp<<<NSEQ, 128, 0, stream>>>(xmid, norm2_w, f1w, fc1_b, f2w, fc2_b, out);
}
